// GRUClassifier_3444563771396
// MI455X (gfx1250) — matrix operand order repaired and checked by reading, not compiled, not run on hardware
//
#include <hip/hip_runtime.h>
#include <cstdint>
#include <cstddef>

// GRU classifier: B=32, T=2048, D=64, H=512, 3H=1536, C=10  (fp32 in/out, bf16 WMMA compute)
#define B_  32
#define T_  2048
#define D_  64
#define H_  512
#define C_  10
#define NWG 16          // persistent workgroups; each owns 32 hidden columns
#define SLICE 32        // hidden cols per WG
#define KC_X 2          // K-chunks (32 wide) for x part (D=64)
#define KC_H 16         // K-chunks for h part (H=512)
#define KC_T 18
#define FRAG_U16 512    // one 16x32(bf16) / 32x16(bf16) fragment = 32 lanes * 16 u16 = 1KB

typedef __attribute__((ext_vector_type(16))) __bf16       v16bf;
typedef __attribute__((ext_vector_type(8)))  float        v8f;
typedef __attribute__((ext_vector_type(8)))  unsigned int v8u;
typedef __attribute__((ext_vector_type(4)))  unsigned int v4u;
typedef __attribute__((ext_vector_type(4)))  int          v4i;

// ---- workspace layout (bytes) ----
static constexpr size_t OFF_CNT  = 0;                                  // barrier counter
static constexpr size_t OFF_WFC  = 1024;                               // w_fc B-frags: 16 kc * 1KB
static constexpr size_t SZ_WFC   = (size_t)KC_H * FRAG_U16 * 2;
static constexpr size_t OFF_WCAT = OFF_WFC + SZ_WFC;                   // [wg][gate][ntile][kc] B-frags
static constexpr size_t SZ_WCAT  = (size_t)NWG * 3 * 2 * KC_T * FRAG_U16 * 2;
static constexpr size_t OFF_XF   = OFF_WCAT + SZ_WCAT;                 // x A-frags [t][mtile][kc]
static constexpr size_t SZ_XF    = (size_t)T_ * 2 * KC_X * FRAG_U16 * 2;
static constexpr size_t OFF_HS   = OFF_XF + SZ_XF;                     // h A-frags [t+1 slots][mtile][kc]
static constexpr size_t HS_SLOT_U16 = (size_t)2 * KC_H * FRAG_U16;     // 16384 u16 = 32KB per step

// ---- CDNA5 async global->LDS copy (ASYNCcnt-tracked, bypasses VGPRs) ----
#ifndef __has_builtin
#define __has_builtin(x) 0
#endif

__device__ __forceinline__ void async_ld16(void* lds_dst, const void* gsrc) {
#if __has_builtin(__builtin_amdgcn_global_load_async_to_lds_b128)
    __builtin_amdgcn_global_load_async_to_lds_b128(
        (__attribute__((address_space(1))) v4i*)(void*)gsrc,
        (__attribute__((address_space(3))) v4i*)lds_dst, 0, 0);
#else
    unsigned           l = (unsigned)(unsigned long long)lds_dst;      // LDS byte address
    unsigned long long g = (unsigned long long)gsrc;                   // 64-bit global address
    asm volatile("global_load_async_to_lds_b128 %0, %1, off"
                 :: "v"(l), "v"(g) : "memory");
#endif
}

__device__ __forceinline__ void wait_async0() {
#if __has_builtin(__builtin_amdgcn_s_wait_asynccnt)
    __builtin_amdgcn_s_wait_asynccnt(0);
#else
    asm volatile("s_wait_asynccnt 0x0" ::: "memory");
#endif
}

// ---- fragment index helpers (CDNA5 ISA 7.12.2, 16-bit matrices, wave32) ----
__device__ __forceinline__ int afrag_pos(int m, int k) {      // A: 16x32 (MxK)
    int lane = m + 16 * ((k >> 3) & 1);
    int idx  = (k & 7) + 8 * (k >> 4);
    return lane * 16 + idx;
}
__device__ __forceinline__ int bfrag_pos(int k, int n) {      // B: 32x16 (KxN)
    int lane = n + 16 * ((k >> 3) & 1);
    int idx  = (k & 7) + 8 * (k >> 4);
    return lane * 16 + idx;
}
__device__ __forceinline__ unsigned short f2bf(float f) {
    unsigned u = __float_as_uint(f);
    u += 0x7fffu + ((u >> 16) & 1u);                          // RNE
    return (unsigned short)(u >> 16);
}
__device__ __forceinline__ float bf2f(unsigned short h) {
    return __uint_as_float(((unsigned)h) << 16);
}
__device__ __forceinline__ v16bf load_frag(const unsigned short* p) {
    return __builtin_bit_cast(v16bf, *(const v8u*)p);
}

// ================= prep kernels =================
__global__ __launch_bounds__(256) void k_init(unsigned int* cnt, unsigned int* hs0_u32) {
    int i = blockIdx.x * 256 + threadIdx.x;
    if (i == 0) *cnt = 0u;
    if (i < (int)(HS_SLOT_U16 / 2)) hs0_u32[i] = 0u;          // h0 = zeros (slot 0)
}

__global__ __launch_bounds__(256) void k_prep_wfc(const float* w_fc, unsigned short* wfc) {
    int i = blockIdx.x * 256 + threadIdx.x;                   // 16*32*16 = 8192
    int n  = i & 15;
    int k  = (i >> 4) & 31;
    int kc = i >> 9;
    float v = (n < C_) ? w_fc[n * H_ + kc * 32 + k] : 0.f;
    wfc[(size_t)kc * FRAG_U16 + bfrag_pos(k, n)] = f2bf(v);
}

__global__ __launch_bounds__(256) void k_prep_wcat(const float* w_ih, const float* w_hh,
                                                   unsigned short* wcat) {
    int i = blockIdx.x * 256 + threadIdx.x;                   // 16*3*2*18*32*16 = 884736
    int n = i & 15;      i >>= 4;
    int k = i & 31;      i >>= 5;
    int kc = i % KC_T;   i /= KC_T;
    int ntile = i & 1;   i >>= 1;
    int gate = i % 3;    i /= 3;
    int wg = i;                                               // 0..15
    int row = gate * H_ + wg * SLICE + ntile * 16 + n;        // gate-major rows (r,z,n)
    float v = (kc < KC_X) ? w_ih[(size_t)row * D_ + kc * 32 + k]
                          : w_hh[(size_t)row * H_ + (kc - KC_X) * 32 + k];
    size_t base = ((((size_t)wg * 3 + gate) * 2 + ntile) * KC_T + kc) * FRAG_U16;
    wcat[base + bfrag_pos(k, n)] = f2bf(v);
}

__global__ __launch_bounds__(256) void k_prep_xf(const float* x, unsigned short* xf) {
    int i = blockIdx.x * 256 + threadIdx.x;                   // T*B*D = 4194304
    int d = i % D_;      i /= D_;
    int b = i % B_;      i /= B_;
    int t = i;
    float v = x[((size_t)b * T_ + t) * D_ + d];
    int mtile = b >> 4, kc = d >> 5;
    size_t base = (((size_t)t * 2 + mtile) * KC_X + kc) * FRAG_U16;
    xf[base + afrag_pos(b & 15, d & 31)] = f2bf(v);
}

// ================= persistent GRU recurrence =================
// 16 WGs x 384 threads (12 waves). Wave (gate,mtile,ntile) owns one 16x16 tile of this
// WG's 32-column gate slice. h exchanged via global A-fragments in hs[t], grid barrier/step.
// W B-fragments are loop-invariant -> compiler keeps them register-resident (verified R1).
__global__ __launch_bounds__(384) void k_gru(const float* b_ih, const float* b_hh,
                                             const unsigned short* __restrict__ wcat,
                                             const unsigned short* __restrict__ xf,
                                             unsigned short* __restrict__ hs,
                                             unsigned int* cnt) {
    __shared__ unsigned short sH[2 * KC_H * FRAG_U16];        // full h A-frags, 32 KB
    __shared__ float sRZ[2 * 2 * 2 * 256];                    // r/z tiles, 8 KB

    const int tid   = threadIdx.x;
    const int lane  = tid & 31;
    const int w     = tid >> 5;              // 0..11
    const int gate  = w >> 2;                // 0:r 1:z 2:n
    const int mtile = (w >> 1) & 1;
    const int ntile = w & 1;
    const int wg    = blockIdx.x;

    const int jcol = wg * SLICE + ntile * 16 + (lane & 15);   // global hidden column of this lane
    const int grow = gate * H_ + jcol;
    const float bih = b_ih[grow];
    const float bhh = b_hh[grow];

    const unsigned short* wf = wcat + ((((size_t)wg * 3 + gate) * 2 + ntile) * KC_T) * FRAG_U16;

    // n-wave scatter geometry (D-frag element -> A-frag position of column jcol)
    const int kc_j = jcol >> 5;
    const int kk_j = jcol & 31;
    const int aidx_j = (kk_j & 7) + 8 * (kk_j >> 4);
    const int alane_base = 16 * ((kk_j >> 3) & 1);

    for (int t = 0; t < T_; ++t) {
        // ---- async-stage h(t) A-fragments into LDS (2048 x 16B, ASYNCcnt-tracked) ----
        const unsigned short* hsrc = hs + (size_t)t * HS_SLOT_U16;
        for (int i = tid; i < (int)(HS_SLOT_U16 / 8); i += 384)
            async_ld16(sH + i * 8, hsrc + i * 8);
        wait_async0();
        __syncthreads();

        // prefetch next timestep's x fragments into cache while we compute
        if (t + 1 < T_)
            __builtin_prefetch(xf + (((size_t)(t + 1) * 2 + mtile) * KC_X) * FRAG_U16 + lane * 16, 0, 0);

        // ---- GEMM: acc_x = x_t @ w_ih^T slice, acc_h = h @ w_hh^T slice ----
        v8f accx = {};
        v8f acch = {};
        const unsigned short* xfr = xf + (((size_t)t * 2 + mtile) * KC_X) * FRAG_U16;
#pragma unroll
        for (int kc = 0; kc < KC_X; ++kc) {
            v16bf a = load_frag(xfr + (size_t)kc * FRAG_U16 + lane * 16);
            v16bf b = load_frag(wf + (size_t)kc * FRAG_U16 + lane * 16);
            accx = __builtin_amdgcn_wmma_f32_16x16x32_bf16(false, a, false, b,
                                                           (short)0, accx, false, false);
        }
#pragma unroll
        for (int kc = 0; kc < KC_H; ++kc) {
            v16bf a = load_frag(sH + (size_t)(mtile * KC_H + kc) * FRAG_U16 + lane * 16);
            v16bf b = load_frag(wf + (size_t)(KC_X + kc) * FRAG_U16 + lane * 16);
            acch = __builtin_amdgcn_wmma_f32_16x16x32_bf16(false, a, false, b,
                                                           (short)0, acch, false, false);
        }

        // ---- gates: r,z waves activate and stage; n wave combines + writes h(t+1) ----
        if (gate < 2) {
            float* st = sRZ + (((gate * 2 + mtile) * 2 + ntile) * 256) + lane * 8;
#pragma unroll
            for (int v = 0; v < 8; ++v) {
                float pre = accx[v] + acch[v] + bih + bhh;
                st[v] = 1.f / (1.f + __expf(-pre));
            }
        }
        __syncthreads();
        if (gate == 2) {
            const float* rr = sRZ + (((0 * 2 + mtile) * 2 + ntile) * 256) + lane * 8;
            const float* zz = sRZ + (((1 * 2 + mtile) * 2 + ntile) * 256) + lane * 8;
            unsigned short* hdst = hs + (size_t)(t + 1) * HS_SLOT_U16
                                      + (size_t)(mtile * KC_H + kc_j) * FRAG_U16;
            const unsigned short* hcur = sH + (size_t)(mtile * KC_H + kc_j) * FRAG_U16;
#pragma unroll
            for (int v = 0; v < 8; ++v) {
                int m = 8 * (lane >> 4) + v;
                int apos = (m + alane_base) * 16 + aidx_j;
                float hp = bf2f(hcur[apos]);
                float nv = tanhf(accx[v] + bih + rr[v] * (acch[v] + bhh));
                float z  = zz[v];
                hdst[apos] = f2bf((1.f - z) * nv + z * hp);
            }
            __threadfence();                                  // make h(t+1) agent-visible
        }
        __syncthreads();

        // ---- grid barrier (monotonic counter; reset each launch by k_init) ----
        if (tid == 0) {
            __hip_atomic_fetch_add(cnt, 1u, __ATOMIC_RELEASE, __HIP_MEMORY_SCOPE_AGENT);
            const unsigned target = (unsigned)(NWG * (t + 1));
            while (__hip_atomic_load(cnt, __ATOMIC_ACQUIRE, __HIP_MEMORY_SCOPE_AGENT) < target)
                __builtin_amdgcn_s_sleep(1);
        }
        __syncthreads();
    }
}

// ================= classifier: out[b,t,c] = h_t @ w_fc^T + b_fc =================
__global__ __launch_bounds__(256) void k_fc(const unsigned short* __restrict__ hs,
                                            const unsigned short* __restrict__ wfc,
                                            const float* b_fc, float* out) {
    __shared__ unsigned short sW[KC_H * FRAG_U16];            // 16 KB
    for (int i = threadIdx.x; i < KC_H * FRAG_U16 / 8; i += 256)
        async_ld16(sW + i * 8, wfc + i * 8);
    wait_async0();
    __syncthreads();

    const int lane = threadIdx.x & 31;
    const int widx = blockIdx.x * 8 + (threadIdx.x >> 5);     // 0..4095
    const int t = widx >> 1;
    const int mtile = widx & 1;

    v8f acc = {};
    const unsigned short* hsrc = hs + (size_t)(t + 1) * HS_SLOT_U16
                                    + (size_t)mtile * KC_H * FRAG_U16;
#pragma unroll
    for (int kc = 0; kc < KC_H; ++kc) {
        v16bf a = load_frag(hsrc + (size_t)kc * FRAG_U16 + lane * 16);
        v16bf b = load_frag(sW + (size_t)kc * FRAG_U16 + lane * 16);
        acc = __builtin_amdgcn_wmma_f32_16x16x32_bf16(false, a, false, b,
                                                      (short)0, acc, false, false);
    }
    const int c = lane & 15;
    if (c < C_) {
        const float bias = b_fc[c];
#pragma unroll
        for (int v = 0; v < 8; ++v) {
            int b = mtile * 16 + 8 * (lane >> 4) + v;
            out[((size_t)b * T_ + t) * C_ + c] = acc[v] + bias;
        }
    }
}

// ================= host launcher =================
extern "C" void kernel_launch(void* const* d_in, const int* in_sizes, int n_in,
                              void* d_out, int out_size, void* d_ws, size_t ws_size,
                              hipStream_t stream) {
    (void)in_sizes; (void)n_in; (void)out_size; (void)ws_size;
    const float* x    = (const float*)d_in[0];
    const float* w_ih = (const float*)d_in[1];
    const float* w_hh = (const float*)d_in[2];
    const float* b_ih = (const float*)d_in[3];
    const float* b_hh = (const float*)d_in[4];
    const float* w_fc = (const float*)d_in[5];
    const float* b_fc = (const float*)d_in[6];
    float* out = (float*)d_out;

    char* ws = (char*)d_ws;
    unsigned int*   cnt  = (unsigned int*)(ws + OFF_CNT);
    unsigned short* wfc  = (unsigned short*)(ws + OFF_WFC);
    unsigned short* wcat = (unsigned short*)(ws + OFF_WCAT);
    unsigned short* xf   = (unsigned short*)(ws + OFF_XF);
    unsigned short* hs   = (unsigned short*)(ws + OFF_HS);

    k_init     <<<32,    256, 0, stream>>>(cnt, (unsigned int*)hs);
    k_prep_wfc <<<32,    256, 0, stream>>>(w_fc, wfc);
    k_prep_wcat<<<3456,  256, 0, stream>>>(w_ih, w_hh, wcat);
    k_prep_xf  <<<16384, 256, 0, stream>>>(x, xf);
    k_gru      <<<NWG,   384, 0, stream>>>(b_ih, b_hh, wcat, xf, hs, cnt);
    k_fc       <<<512,   256, 0, stream>>>(hs, wfc, b_fc, out);
}
